// Net_62629213110506
// MI455X (gfx1250) — matrix operand order repaired and checked by reading, not compiled, not run on hardware
//
#include <hip/hip_runtime.h>
#include <hip/hip_bf16.h>

// ---------------------------------------------------------------------------
// MI455X (gfx1250, wave32) implementation.
// GRU stack dominated by GEMMs -> v_wmma_f32_16x16x32_bf16,
// B tiles double-buffered in LDS via global_load_async_to_lds_b128 (ASYNCcnt).
// ---------------------------------------------------------------------------

typedef __attribute__((ext_vector_type(8)))  __bf16 v8bf;
typedef __attribute__((ext_vector_type(16))) __bf16 v16bf;
typedef __attribute__((ext_vector_type(8)))  float  v8f;

union BF16x16 { v16bf v; v8bf h[2]; };

#define T_STEPS 8
#define BATCH   4096
#define HSZ     512
#define N3      1536      // 3*H
#define KIN     1096      // 96 timing + 1000 feats
#define KPAD    1120      // KIN padded to multiple of 32

__device__ __forceinline__ unsigned short f2bf(float f) {
  unsigned int u = __float_as_uint(f);
  u += 0x7FFFu + ((u >> 16) & 1u);              // round-to-nearest-even
  return (unsigned short)(u >> 16);
}
__device__ __forceinline__ float eluf(float v) { return v > 0.f ? v : expm1f(v); }
__device__ __forceinline__ float sigm(float v) { return 1.f / (1.f + expf(-v)); }

// ---------------------------------------------------------------------------
// Generic utility kernels
// ---------------------------------------------------------------------------
__global__ void zero_u32(unsigned int* __restrict__ p, long long n) {
  long long i = (long long)blockIdx.x * 256 + threadIdx.x;
  long long stride = (long long)gridDim.x * 256;
  for (; i < n; i += stride) p[i] = 0u;
}

__global__ void cvt_bf(const float* __restrict__ s, unsigned short* __restrict__ d,
                       long long n) {
  long long i = (long long)blockIdx.x * 256 + threadIdx.x;
  long long stride = (long long)gridDim.x * 256;
  for (; i < n; i += stride) d[i] = f2bf(s[i]);
}

// gru1_Wih (1536 x 1096) -> bf16, K padded to 1120 with zeros
__global__ void pad_wih1(const float* __restrict__ w, unsigned short* __restrict__ d) {
  long long total = (long long)N3 * KPAD;
  long long stride = (long long)gridDim.x * 256;
  for (long long i = (long long)blockIdx.x * 256 + threadIdx.x; i < total; i += stride) {
    int n = (int)(i / KPAD), k = (int)(i % KPAD);
    d[i] = f2bf(k < KIN ? w[(size_t)n * KIN + k] : 0.f);
  }
}

// ---------------------------------------------------------------------------
// Conv / BatchNorm front-end.
// bn param block layout (floats): [0..4]=a1 [8..12]=b1 [16..23]=a2 [24..31]=b2
// ---------------------------------------------------------------------------
__device__ __forceinline__ void x1b_at(const float* __restrict__ xr, int r, int c,
                                       const float* __restrict__ cw,
                                       const float* __restrict__ cb,
                                       const float* __restrict__ bn, float out[5]) {
  float xv = xr[r * 5 + c];
  out[0] = xv * bn[0] + bn[8];
  #pragma unroll
  for (int ch = 0; ch < 4; ++ch)
    out[ch + 1] = eluf(cw[ch] * xv + cb[ch]) * bn[1 + ch] + bn[9 + ch];
}

__device__ __forceinline__ void conv2_at(const float* __restrict__ xr, int r, int c,
                                         const float* __restrict__ cw,
                                         const float* __restrict__ cb,
                                         const float* __restrict__ bn,
                                         const float* __restrict__ c2w,
                                         const float* __restrict__ c2b, float out[3]) {
  float acc[3] = {c2b[0], c2b[1], c2b[2]};
  #pragma unroll
  for (int dy = -1; dy <= 1; ++dy) {
    #pragma unroll
    for (int dx = -1; dx <= 1; ++dx) {
      int rr = r + dy, cc = c + dx;
      if (rr < 0 || rr >= 25 || cc < 0 || cc >= 5) continue;
      float v5[5];
      x1b_at(xr, rr, cc, cw, cb, bn, v5);
      #pragma unroll
      for (int o = 0; o < 3; ++o) {
        #pragma unroll
        for (int ci = 0; ci < 5; ++ci)
          acc[o] += c2w[((o * 5 + ci) * 3 + (dy + 1)) * 3 + (dx + 1)] * v5[ci];
      }
    }
  }
  #pragma unroll
  for (int o = 0; o < 3; ++o) out[o] = eluf(acc[o]);
}

__global__ void bn1_stats(const float* __restrict__ x, const float* __restrict__ cw,
                          const float* __restrict__ cb, float* __restrict__ sums) {
  __shared__ float sred[10 * 256];
  float ls[10];
  #pragma unroll
  for (int i = 0; i < 10; ++i) ls[i] = 0.f;
  long long n = (long long)(T_STEPS * BATCH) * 125;
  long long stride = (long long)gridDim.x * 256;
  for (long long i = (long long)blockIdx.x * 256 + threadIdx.x; i < n; i += stride) {
    float xv = x[i];
    float v[5];
    v[0] = xv;
    #pragma unroll
    for (int c = 0; c < 4; ++c) v[c + 1] = eluf(cw[c] * xv + cb[c]);
    #pragma unroll
    for (int c = 0; c < 5; ++c) { ls[c] += v[c]; ls[5 + c] += v[c] * v[c]; }
  }
  int tid = threadIdx.x;
  #pragma unroll
  for (int c = 0; c < 10; ++c) sred[c * 256 + tid] = ls[c];
  for (int s = 128; s > 0; s >>= 1) {
    __syncthreads();
    if (tid < s) {
      #pragma unroll
      for (int c = 0; c < 10; ++c) sred[c * 256 + tid] += sred[c * 256 + tid + s];
    }
  }
  if (tid == 0) {
    #pragma unroll
    for (int c = 0; c < 10; ++c) atomicAdd(&sums[c], sred[c * 256]);
  }
}

__global__ void bn1_final(const float* __restrict__ sums, const float* __restrict__ g,
                          const float* __restrict__ b, float* __restrict__ bn) {
  int c = threadIdx.x;
  if (c < 5) {
    float cnt = (float)((long long)T_STEPS * BATCH * 125);
    float mean = sums[c] / cnt;
    float var = sums[5 + c] / cnt - mean * mean;
    float rstd = rsqrtf(var + 1e-5f);
    bn[c] = rstd * g[c];
    bn[8 + c] = b[c] - mean * rstd * g[c];
  }
}

__global__ void bn2_stats(const float* __restrict__ x, const float* __restrict__ cw,
                          const float* __restrict__ cb, const float* __restrict__ c2w,
                          const float* __restrict__ c2b, const float* __restrict__ bn,
                          float* __restrict__ sums2) {
  __shared__ float sred[16 * 256];
  float ls[16];
  #pragma unroll
  for (int i = 0; i < 16; ++i) ls[i] = 0.f;
  long long n = (long long)(T_STEPS * BATCH) * 125;
  long long stride = (long long)gridDim.x * 256;
  for (long long i = (long long)blockIdx.x * 256 + threadIdx.x; i < n; i += stride) {
    long long row = i / 125;
    int p = (int)(i % 125), r = p / 5, c = p % 5;
    const float* xr = x + row * 125;
    float v5[5], o3[3];
    x1b_at(xr, r, c, cw, cb, bn, v5);
    conv2_at(xr, r, c, cw, cb, bn, c2w, c2b, o3);
    #pragma unroll
    for (int ch = 0; ch < 5; ++ch) { ls[ch] += v5[ch]; ls[8 + ch] += v5[ch] * v5[ch]; }
    #pragma unroll
    for (int ch = 0; ch < 3; ++ch) { ls[5 + ch] += o3[ch]; ls[13 + ch] += o3[ch] * o3[ch]; }
  }
  int tid = threadIdx.x;
  #pragma unroll
  for (int c = 0; c < 16; ++c) sred[c * 256 + tid] = ls[c];
  for (int s = 128; s > 0; s >>= 1) {
    __syncthreads();
    if (tid < s) {
      #pragma unroll
      for (int c = 0; c < 16; ++c) sred[c * 256 + tid] += sred[c * 256 + tid + s];
    }
  }
  if (tid == 0) {
    #pragma unroll
    for (int c = 0; c < 16; ++c) atomicAdd(&sums2[c], sred[c * 256]);
  }
}

__global__ void bn2_final(const float* __restrict__ sums2, const float* __restrict__ g,
                          const float* __restrict__ b, float* __restrict__ bn) {
  int c = threadIdx.x;
  if (c < 8) {
    float cnt = (float)((long long)T_STEPS * BATCH * 125);
    float mean = sums2[c] / cnt;
    float var = sums2[8 + c] / cnt - mean * mean;
    float rstd = rsqrtf(var + 1e-5f);
    bn[16 + c] = rstd * g[c];
    bn[24 + c] = b[c] - mean * rstd * g[c];
  }
}

// Assemble combined[t] as bf16, K padded to KPAD with zeros.
__global__ void build_combined(const float* __restrict__ x, const float* __restrict__ timing,
                               const float* __restrict__ cw, const float* __restrict__ cb,
                               const float* __restrict__ c2w, const float* __restrict__ c2b,
                               const float* __restrict__ bn, int t,
                               unsigned short* __restrict__ cmb) {
  long long total = (long long)BATCH * KPAD;
  long long stride = (long long)gridDim.x * 256;
  for (long long i = (long long)blockIdx.x * 256 + threadIdx.x; i < total; i += stride) {
    int row = (int)(i / KPAD), col = (int)(i % KPAD);
    long long grow = (long long)t * BATCH + row;
    float v;
    if (col < 96) {
      v = timing[grow * 96 + col];
    } else if (col < 1096) {
      int f = col - 96, c = f / 125, p = f % 125, r = p / 5, cc = p % 5;
      const float* xr = x + grow * 125;
      float val;
      if (c < 5) { float v5[5]; x1b_at(xr, r, cc, cw, cb, bn, v5); val = v5[c]; }
      else { float o3[3]; conv2_at(xr, r, cc, cw, cb, bn, c2w, c2b, o3); val = o3[c - 5]; }
      v = val * bn[16 + c] + bn[24 + c];
    } else {
      v = 0.f;
    }
    cmb[i] = f2bf(v);
  }
}

// ---------------------------------------------------------------------------
// WMMA bf16 GEMM:  C[M x N] = A[M x K](bf16) * W[N x K]^T(bf16) + bias[N]
// grid = (N/64, M/128), block = 128 threads (4 waves).
// Each wave: 32(M) x 64(N) -> acc[2][4], 8 WMMAs per K-step of 32.
// W tiles DOUBLE-BUFFERED in LDS via global_load_async_to_lds_b128:
// tile i+1 streams in (ASYNCcnt) while tile i feeds the WMMAs; one barrier
// per K-step (dscnt waits before the WMMAs guarantee prior-tile LDS reads
// have landed in VGPRs before any wave reaches the next barrier).
// ---------------------------------------------------------------------------
__global__ __launch_bounds__(128)
void wmma_gemm_bf16(const unsigned short* __restrict__ A, int lda,
                    const unsigned short* __restrict__ W,
                    const float* __restrict__ bias,
                    float* __restrict__ C, int ldc, int K) {
  __shared__ __align__(16) unsigned short tileB[2][64 * 32];   // 2 x 4KB

  const int tid  = threadIdx.x;
  const int lane = tid & 31;
  const int wave = tid >> 5;
  const int half = lane >> 4;     // 0: lanes 0-15, 1: lanes 16-31
  const int l15  = lane & 15;
  const int blockN = blockIdx.x * 64;
  const int blockM = blockIdx.y * 128;

  const size_t rowA0 = (size_t)(blockM + wave * 32 + l15);
  const unsigned short* aBase0 = A + rowA0 * (size_t)lda;
  const unsigned short* aBase1 = aBase0 + (size_t)16 * lda;

  // per-thread async-copy lds dst / global src (2x 16B chunks per thread)
  const int cr0 = tid >> 1, cc0 = (tid & 1) * 2;
  const unsigned ldsDst0 = (unsigned)(uintptr_t)(&tileB[0][cr0 * 32 + cc0 * 8]);
  const unsigned ldsDst1 = ldsDst0 + 16;
  const unsigned short* gSrc0 = W + (size_t)(blockN + cr0) * (size_t)K + cc0 * 8;

  v8f acc[2][4] = {};
  const int nk = K >> 5;

  // prologue: stream tile 0 into buffer 0
  {
    const unsigned short* g0 = gSrc0;
    asm volatile("global_load_async_to_lds_b128 %0, %1, off"
                 :: "v"(ldsDst0), "v"(g0) : "memory");
    asm volatile("global_load_async_to_lds_b128 %0, %1, off"
                 :: "v"(ldsDst1), "v"(g0 + 8) : "memory");
  }

  for (int i = 0; i < nk; ++i) {
    const int k0 = i << 5;
    const unsigned bufSel = (unsigned)(i & 1);

    // our tile-i copy complete; after barrier: everyone's tile-i copy complete,
    // and everyone's tile-(i-1) LDS reads already landed in VGPRs.
    asm volatile("s_wait_asynccnt 0x0" ::: "memory");
    __syncthreads();

    // stream tile i+1 into the alternate buffer while we compute on tile i
    if (i + 1 < nk) {
      const unsigned nb = ((i + 1) & 1) ? 4096u : 0u;
      const unsigned short* g0 = gSrc0 + ((i + 1) << 5);
      asm volatile("global_load_async_to_lds_b128 %0, %1, off"
                   :: "v"(ldsDst0 + nb), "v"(g0) : "memory");
      asm volatile("global_load_async_to_lds_b128 %0, %1, off"
                   :: "v"(ldsDst1 + nb), "v"(g0 + 8) : "memory");
    }

    // A fragments (16-bit A 16x32 layout: lanes 0-15 K 0-7,16-23; lanes 16-31 K 8-15,24-31)
    BF16x16 a0, a1;
    a0.h[0] = *reinterpret_cast<const v8bf*>(aBase0 + k0 + half * 8);
    a0.h[1] = *reinterpret_cast<const v8bf*>(aBase0 + k0 + 16 + half * 8);
    a1.h[0] = *reinterpret_cast<const v8bf*>(aBase1 + k0 + half * 8);
    a1.h[1] = *reinterpret_cast<const v8bf*>(aBase1 + k0 + 16 + half * 8);
    if (k0 + 32 < K) {
      __builtin_prefetch(aBase0 + k0 + 32, 0, 1);
      __builtin_prefetch(aBase1 + k0 + 32, 0, 1);
    }

    // B fragments from LDS tile i, then 8 WMMAs
    const unsigned short* tb = &tileB[bufSel][0];
    BF16x16 bfr[4];
    #pragma unroll
    for (int j = 0; j < 4; ++j) {
      const unsigned short* bb = tb + (j * 16 + l15) * 32 + half * 8;
      bfr[j].h[0] = *reinterpret_cast<const v8bf*>(bb);
      bfr[j].h[1] = *reinterpret_cast<const v8bf*>(bb + 16);
    }
    #pragma unroll
    for (int j = 0; j < 4; ++j) {
      acc[0][j] = __builtin_amdgcn_wmma_f32_16x16x32_bf16(
          false, a0.v, false, bfr[j].v, (short)0, acc[0][j], false, false);
      acc[1][j] = __builtin_amdgcn_wmma_f32_16x16x32_bf16(
          false, a1.v, false, bfr[j].v, (short)0, acc[1][j], false, false);
    }
  }

  // Epilogue. f32 C 16x16 layout: VGPR r -> M = r + 8*half, N = l15.
  #pragma unroll
  for (int mi = 0; mi < 2; ++mi) {
    const int mrow = blockM + wave * 32 + mi * 16 + half * 8;
    #pragma unroll
    for (int j = 0; j < 4; ++j) {
      int n = blockN + j * 16 + l15;
      float bv = bias ? bias[n] : 0.f;
      #pragma unroll
      for (int r = 0; r < 8; ++r)
        C[(size_t)(mrow + r) * (size_t)ldc + n] = acc[mi][j][r] + bv;
    }
  }
}

// ---------------------------------------------------------------------------
// GRU cell (elementwise), residual add, writes f32 state + bf16 copy
// ---------------------------------------------------------------------------
__global__ void gru_cell(const float* __restrict__ gi, const float* __restrict__ gh,
                         const float* __restrict__ hprev, const float* __restrict__ res,
                         float flag, float* __restrict__ hout,
                         unsigned short* __restrict__ hbf) {
  int i = blockIdx.x * 256 + threadIdx.x;      // BATCH*HSZ exact
  int b = i >> 9, j = i & 511;
  const float* gib = gi + (size_t)b * N3;
  const float* ghb = gh + (size_t)b * N3;
  float r = sigm(gib[j] + ghb[j]);
  float z = sigm(gib[512 + j] + ghb[512 + j]);
  float n = tanhf(gib[1024 + j] + r * ghb[1024 + j]);
  float h = (1.f - z) * n + z * hprev[i];
  if (res) h += flag * res[i];
  hout[i] = h;
  hbf[i] = f2bf(h);
}

// ---------------------------------------------------------------------------
// NALU head
// ---------------------------------------------------------------------------
__global__ void logabs_k(const float* __restrict__ h, unsigned short* __restrict__ lbf) {
  int i = blockIdx.x * 256 + threadIdx.x;      // BATCH*HSZ exact
  lbf[i] = f2bf(logf(fabsf(h[i]) + 1e-10f));
}

// Wt[n*512+k] = tanh(What[k,n])*sigmoid(Mhat[k,n]); Gt likewise; n>=125 -> 0
__global__ void nalu_prep(const float* __restrict__ What, const float* __restrict__ Mhat,
                          const float* __restrict__ G, unsigned short* __restrict__ Wt,
                          unsigned short* __restrict__ Gt) {
  int i = blockIdx.x * 256 + threadIdx.x;      // 128*512 exact
  int n = i >> 9, k = i & 511;
  float w = 0.f, g = 0.f;
  if (n < 125) {
    w = tanhf(What[k * 125 + n]) * sigm(Mhat[k * 125 + n]);
    g = G[k * 125 + n];
  }
  Wt[i] = f2bf(w);
  Gt[i] = f2bf(g);
}

__global__ void nalu_out(const float* __restrict__ Ab, const float* __restrict__ Mb,
                         const float* __restrict__ Gb, float* __restrict__ out) {
  int i = blockIdx.x * 256 + threadIdx.x;      // BATCH*125 = 512000 exact
  int b = i / 125, o = i % 125;
  float a = Ab[b * 128 + o];
  float m = expf(Mb[b * 128 + o]);
  float g = sigm(Gb[b * 128 + o]);
  out[i] = g * a + (1.f - g) * m;
}

// ---------------------------------------------------------------------------
// Host orchestration
// ---------------------------------------------------------------------------
extern "C" void kernel_launch(void* const* d_in, const int* in_sizes, int n_in,
                              void* d_out, int out_size, void* d_ws, size_t ws_size,
                              hipStream_t stream) {
  const float* x       = (const float*)d_in[0];
  const float* timing  = (const float*)d_in[1];
  const float* conv1_w = (const float*)d_in[2];
  const float* conv1_b = (const float*)d_in[3];
  const float* bn1_g   = (const float*)d_in[4];
  const float* bn1_b   = (const float*)d_in[5];
  const float* conv2_w = (const float*)d_in[6];
  const float* conv2_b = (const float*)d_in[7];
  const float* bn2_g   = (const float*)d_in[8];
  const float* bn2_b   = (const float*)d_in[9];
  const float* g1_Wih  = (const float*)d_in[10];
  const float* g1_Whh  = (const float*)d_in[11];
  const float* g1_bih  = (const float*)d_in[12];
  const float* g1_bhh  = (const float*)d_in[13];
  const float* gs_Wih  = (const float*)d_in[14];
  const float* gs_Whh  = (const float*)d_in[15];
  const float* gs_bih  = (const float*)d_in[16];
  const float* gs_bhh  = (const float*)d_in[17];
  const float* What    = (const float*)d_in[18];
  const float* Mhat    = (const float*)d_in[19];
  const float* G       = (const float*)d_in[20];
  float* out = (float*)d_out;

  char* ws = (char*)d_ws;
  size_t off = 0;
  auto alloc = [&](size_t bytes) -> char* {
    char* p = ws + off;
    off = (off + bytes + 255) & ~(size_t)255;
    return p;
  };

  float*          stats  = (float*)alloc(64 * sizeof(float));
  float*          bnP    = (float*)alloc(32 * sizeof(float));
  unsigned short* wih1   = (unsigned short*)alloc((size_t)N3 * KPAD * 2);
  unsigned short* whh1   = (unsigned short*)alloc((size_t)N3 * HSZ * 2);
  unsigned short* wihs   = (unsigned short*)alloc((size_t)5 * N3 * HSZ * 2);
  unsigned short* whhs   = (unsigned short*)alloc((size_t)5 * N3 * HSZ * 2);
  unsigned short* Wt     = (unsigned short*)alloc((size_t)128 * HSZ * 2);
  unsigned short* Gt     = (unsigned short*)alloc((size_t)128 * HSZ * 2);
  unsigned short* cmb    = (unsigned short*)alloc((size_t)BATCH * KPAD * 2);
  float*          gi     = (float*)alloc((size_t)BATCH * N3 * 4);
  float*          gh     = (float*)alloc((size_t)BATCH * N3 * 4);
  float*          hf32   = (float*)alloc((size_t)6 * BATCH * HSZ * 4);
  unsigned short* hbf    = (unsigned short*)alloc((size_t)6 * BATCH * HSZ * 2);
  unsigned short* logbf  = (unsigned short*)alloc((size_t)BATCH * HSZ * 2);
  float*          Abuf   = (float*)alloc((size_t)BATCH * 128 * 4);
  float*          Mbuf   = (float*)alloc((size_t)BATCH * 128 * 4);
  float*          Gbuf   = (float*)alloc((size_t)BATCH * 128 * 4);

  float* hF[6];
  unsigned short* hB[6];
  for (int l = 0; l < 6; ++l) {
    hF[l] = hf32 + (size_t)l * BATCH * HSZ;
    hB[l] = hbf + (size_t)l * BATCH * HSZ;
  }

  // ---- init / weight conversion ----
  zero_u32<<<1, 64, 0, stream>>>((unsigned int*)stats, 64);
  zero_u32<<<4096, 256, 0, stream>>>((unsigned int*)hf32, (long long)6 * BATCH * HSZ);
  zero_u32<<<4096, 256, 0, stream>>>((unsigned int*)hbf, (long long)6 * BATCH * HSZ / 2);
  pad_wih1<<<2048, 256, 0, stream>>>(g1_Wih, wih1);
  cvt_bf<<<1024, 256, 0, stream>>>(g1_Whh, whh1, (long long)N3 * HSZ);
  cvt_bf<<<2048, 256, 0, stream>>>(gs_Wih, wihs, (long long)5 * N3 * HSZ);
  cvt_bf<<<2048, 256, 0, stream>>>(gs_Whh, whhs, (long long)5 * N3 * HSZ);
  nalu_prep<<<256, 256, 0, stream>>>(What, Mhat, G, Wt, Gt);

  // ---- batch-norm statistics (training-mode BN over full batch) ----
  bn1_stats<<<1024, 256, 0, stream>>>(x, conv1_w, conv1_b, stats);
  bn1_final<<<1, 32, 0, stream>>>(stats, bn1_g, bn1_b, bnP);
  bn2_stats<<<1024, 256, 0, stream>>>(x, conv1_w, conv1_b, conv2_w, conv2_b, bnP, stats + 16);
  bn2_final<<<1, 32, 0, stream>>>(stats + 16, bn2_g, bn2_b, bnP);

  auto gemm = [&](const unsigned short* A, int lda, const unsigned short* Wm,
                  const float* bias, float* C, int ldc, int K, int Nn, int M) {
    dim3 grid(Nn / 64, M / 128);
    wmma_gemm_bf16<<<grid, 128, 0, stream>>>(A, lda, Wm, bias, C, ldc, K);
  };

  const int cellBlocks = BATCH * HSZ / 256;   // 8192

  // ---- GRU scan over T timesteps ----
  for (int t = 0; t < T_STEPS; ++t) {
    build_combined<<<4096, 256, 0, stream>>>(x, timing, conv1_w, conv1_b,
                                             conv2_w, conv2_b, bnP, t, cmb);
    // layer 1
    gemm(cmb, KPAD, wih1, g1_bih, gi, N3, KPAD, N3, BATCH);
    gemm(hB[0], HSZ, whh1, g1_bhh, gh, N3, HSZ, N3, BATCH);
    gru_cell<<<cellBlocks, 256, 0, stream>>>(gi, gh, hF[0], nullptr, 0.f, hF[0], hB[0]);

    float flag = (t < T_STEPS - 1) ? 1.f : 0.f;
    // layers 2..6 (grus index j = 0..4); residual on layers 2..5
    for (int j = 0; j < 5; ++j) {
      gemm(hB[j], HSZ, wihs + (size_t)j * N3 * HSZ, gs_bih + j * N3, gi, N3, HSZ, N3, BATCH);
      gemm(hB[j + 1], HSZ, whhs + (size_t)j * N3 * HSZ, gs_bhh + j * N3, gh, N3, HSZ, N3, BATCH);
      const float* res = (j < 4) ? hF[j] : nullptr;
      gru_cell<<<cellBlocks, 256, 0, stream>>>(gi, gh, hF[j + 1], res, flag,
                                               hF[j + 1], hB[j + 1]);
    }
  }

  // ---- NALU head ----
  logabs_k<<<cellBlocks, 256, 0, stream>>>(hF[5], logbf);
  gemm(hB[5], HSZ, Wt, nullptr, Abuf, 128, HSZ, 128, BATCH);
  gemm(logbf, HSZ, Wt, nullptr, Mbuf, 128, HSZ, 128, BATCH);
  gemm(hB[5], HSZ, Gt, nullptr, Gbuf, 128, HSZ, 128, BATCH);
  nalu_out<<<2000, 256, 0, stream>>>(Abuf, Mbuf, Gbuf, out);

  (void)in_sizes; (void)n_in; (void)out_size; (void)ws_size;
}
